// SepReformer_5162550690638
// MI455X (gfx1250) — matrix operand order repaired and checked by reading, not compiled, not run on hardware
//
#include <hip/hip_runtime.h>

// ---------------------------------------------------------------------------
// SepReformer block for MI455X (gfx1250, wave32, WMMA bf16 16x16x32, f32 acc)
// Async global->LDS double-buffered GEMM pipeline (ASYNCcnt), prefetch, WMMA.
// ---------------------------------------------------------------------------

typedef __bf16 bf16_t;
typedef __attribute__((ext_vector_type(8)))  __bf16 v8bf;
typedef __attribute__((ext_vector_type(16))) __bf16 v16bf;
typedef __attribute__((ext_vector_type(8)))  float  v8f;

#define DEV __device__ __forceinline__

static constexpr int  KB  = 4;      // batch
static constexpr int  KC  = 512;    // channels
static constexpr int  KT  = 8192;   // time
static constexpr int  KD  = 1024;   // down
static constexpr int  KH  = 8;      // heads
static constexpr int  KDK = 64;     // head dim
static constexpr int  K6C = 3072;
static constexpr int  K3C = 1536;

enum EpiMode {
  EPI_BF16 = 0,            // bf16 out = acc
  EPI_BIAS_BF16 = 1,       // bf16 out = acc + bias[n]
  EPI_ACC_F32 = 3,         // f32 out += acc
  EPI_BIAS_LS_F32 = 4,     // f32 out = (acc + bias[n]) * scale[n]
  EPI_GATE = 5,            // f32 out = res + sigmoid(acc+bias)*oup(upsample)
  EPI_RES_BIAS_LS_F32 = 6  // f32 out = res + (acc + bias[n]) * scale[n]
};

DEV bf16_t f2bf(float f) {
  union { float f; unsigned u; } c; c.f = f;
  unsigned r = (c.u + 0x7FFFu + ((c.u >> 16) & 1u)) >> 16;
  unsigned short h = (unsigned short)r;
  bf16_t o; __builtin_memcpy(&o, &h, 2);
  return o;
}
DEV float bf2f(bf16_t b) {
  unsigned short h; __builtin_memcpy(&h, &b, 2);
  union { unsigned u; float f; } c; c.u = ((unsigned)h) << 16;
  return c.f;
}
DEV float sigmoidf_(float x) { return 1.f / (1.f + __expf(-x)); }

DEV float block_sum(float v, float* red, int tid) {
  red[tid] = v; __syncthreads();
  for (int s = 128; s > 0; s >>= 1) { if (tid < s) red[tid] += red[tid + s]; __syncthreads(); }
  float r = red[0]; __syncthreads();
  return r;
}
DEV float block_max(float v, float* red, int tid) {
  red[tid] = v; __syncthreads();
  for (int s = 128; s > 0; s >>= 1) { if (tid < s) red[tid] = fmaxf(red[tid], red[tid + s]); __syncthreads(); }
  float r = red[0]; __syncthreads();
  return r;
}

// Async global->LDS (CDNA5 Tensor/Async path, tracked with ASYNCcnt).
// vdst = wave-relative LDS byte offset, vaddr = 64-bit global address.
DEV void async_copy_b128(const void* gptr, void* lptr) {
  unsigned lds = (unsigned)(uintptr_t)lptr;
  unsigned long long ga = (unsigned long long)(uintptr_t)gptr;
  asm volatile("global_load_async_to_lds_b128 %0, %1, off" :: "v"(lds), "v"(ga) : "memory");
}
DEV void async_wait0() { asm volatile("s_wait_asynccnt 0" ::: "memory"); }

// ------------------------------ fp32 -> bf16 -------------------------------
__global__ __launch_bounds__(256)
void cvt_bf16_kernel(const float* __restrict__ in, bf16_t* __restrict__ out, long long n) {
  long long i = (long long)blockIdx.x * 256 + threadIdx.x;
  long long st = (long long)gridDim.x * 256;
  for (; i < n; i += st) out[i] = f2bf(in[i]);
}

// --------------------------- tiled transpose (f32) -------------------------
// in (B, R, S) -> out (B, S, R)
__global__ __launch_bounds__(256)
void transpose_kernel(const float* __restrict__ in, float* __restrict__ out, int R, int S) {
  __shared__ float tile[32][33];
  long long base = (long long)blockIdx.z * R * S;
  int s0 = blockIdx.x * 32, r0 = blockIdx.y * 32;
  int tx = threadIdx.x, ty = threadIdx.y;
  for (int i = ty; i < 32; i += 8)
    tile[i][tx] = in[base + (long long)(r0 + i) * S + s0 + tx];
  __syncthreads();
  for (int i = ty; i < 32; i += 8)
    out[base + (long long)(s0 + i) * R + r0 + tx] = tile[tx][i];
}

// ---------------- avg-pool (r=8) over xt rows + LayerNorm -> bf16 ----------
__global__ __launch_bounds__(256)
void pool_ln_kernel(const float* __restrict__ xt, const float* __restrict__ gam,
                    const float* __restrict__ bet, bf16_t* __restrict__ out) {
  __shared__ float red[256];
  long long row = blockIdx.x;              // b*KD + s
  long long bb = row >> 10, s = row & 1023;
  const float* base = xt + (bb * KT + s * 8) * KC;
  int tid = threadIdx.x;
  float a0 = 0.f, a1 = 0.f;
#pragma unroll
  for (int j = 0; j < 8; ++j) {
    a0 += base[(long long)j * KC + tid];
    a1 += base[(long long)j * KC + tid + 256];
  }
  a0 *= 0.125f; a1 *= 0.125f;
  float mean = block_sum(a0 + a1, red, tid) * (1.f / KC);
  float d0 = a0 - mean, d1 = a1 - mean;
  float var = block_sum(d0 * d0 + d1 * d1, red, tid) * (1.f / KC);
  float inv = rsqrtf(var + 1e-5f);
  bf16_t* o = out + row * KC;
  o[tid]       = f2bf(d0 * inv * gam[tid]       + bet[tid]);
  o[tid + 256] = f2bf(d1 * inv * gam[tid + 256] + bet[tid + 256]);
}

// ------------------------- row-wise LayerNorm -> bf16 ----------------------
__global__ __launch_bounds__(256)
void rowln_kernel(const float* __restrict__ in, const float* __restrict__ gam,
                  const float* __restrict__ bet, bf16_t* __restrict__ out) {
  __shared__ float red[256];
  long long row = blockIdx.x;
  const float* p = in + row * KC;
  int tid = threadIdx.x;
  float a0 = p[tid], a1 = p[tid + 256];
  float mean = block_sum(a0 + a1, red, tid) * (1.f / KC);
  float d0 = a0 - mean, d1 = a1 - mean;
  float var = block_sum(d0 * d0 + d1 * d1, red, tid) * (1.f / KC);
  float inv = rsqrtf(var + 1e-5f);
  bf16_t* o = out + row * KC;
  o[tid]       = f2bf(d0 * inv * gam[tid]       + bet[tid]);
  o[tid + 256] = f2bf(d1 * inv * gam[tid + 256] + bet[tid + 256]);
}

// ------- softmax over 1024 cols, scale 1/8; write bf16 in-place ------------
__global__ __launch_bounds__(256)
void softmax_kernel(float* __restrict__ scores) {
  __shared__ float red[256];
  long long row = blockIdx.x;              // bh*1024 + s
  float* p = scores + row * 1024;
  int tid = threadIdx.x;
  float v[4];
#pragma unroll
  for (int k = 0; k < 4; ++k) v[k] = p[tid + k * 256];
  float mx = fmaxf(fmaxf(v[0], v[1]), fmaxf(v[2], v[3]));
  mx = block_max(mx, red, tid);
  float e[4], sum = 0.f;
#pragma unroll
  for (int k = 0; k < 4; ++k) { e[k] = __expf((v[k] - mx) * 0.125f); sum += e[k]; }
  sum = block_sum(sum, red, tid);
  float inv = 1.f / sum;
  bf16_t* o = (bf16_t*)p;                  // first half of the f32 row
#pragma unroll
  for (int k = 0; k < 4; ++k) o[tid + k * 256] = f2bf(e[k] * inv);
}

// ------- depthwise conv(k=3) + bias + GLU -> g bf16 (8-wide vectorized) ----
__global__ __launch_bounds__(256)
void conv_glu_kernel(const bf16_t* __restrict__ z, const float* __restrict__ dww,
                     const float* __restrict__ dwb, bf16_t* __restrict__ g) {
  constexpr int JB = K3C / 8;              // 192 channel-octets
  long long total = (long long)KB * KT * JB;
  long long i = (long long)blockIdx.x * 256 + threadIdx.x;
  if (i >= total) return;
  int jc = (int)(i % JB);
  long long bt = i / JB;
  int t  = (int)(bt % KT);
  int bb = (int)(bt / KT);
  int j0 = jc * 8;
  float a[8], bg[8];
#pragma unroll
  for (int u = 0; u < 8; ++u) { a[u] = dwb[j0 + u]; bg[u] = dwb[j0 + u + K3C]; }
#pragma unroll
  for (int k = 0; k < 3; ++k) {
    int tt = t + k - 1;
    if (tt < 0 || tt >= KT) continue;
    long long base = ((long long)bb * KT + tt) * K6C;
    v8bf za = *(const v8bf*)&z[base + j0];
    v8bf zb = *(const v8bf*)&z[base + j0 + K3C];
#pragma unroll
    for (int u = 0; u < 8; ++u) {
      a[u]  += bf2f(za[u]) * dww[(j0 + u) * 3 + k];
      bg[u] += bf2f(zb[u]) * dww[(j0 + u + K3C) * 3 + k];
    }
  }
  v8bf r;
#pragma unroll
  for (int u = 0; u < 8; ++u) r[u] = f2bf(a[u] * sigmoidf_(bg[u]));
  *(v8bf*)&g[i * 8] = r;
}

// -------------------- WMMA fragment helpers (gfx1250 layouts) --------------
DEV v16bf load_a_frag(const bf16_t* p /* row base + lkA */) {
  v8bf lo = *(const v8bf*)p;
  v8bf hi = *(const v8bf*)(p + 8);
  return __builtin_shufflevector(lo, hi, 0,1,2,3,4,5,6,7,8,9,10,11,12,13,14,15);
}

// ---------------------------------------------------------------------------
// Generic batched GEMM:  C[z][m][n] (+)= sum_k A[z][m][k] * W[z][n or k][k or n]
// z = z1*NZ2 + z2, addressing base + z1*s1 + z2*s2 (element strides).
// WKMAJ=true : W layout (N,K), async double-buffered global->LDS pipeline.
// WKMAJ=false: W layout (K,N), synchronous gather staging.
// ---------------------------------------------------------------------------
template<int BM, int BN, int BK, int WM, int WN, int WTM, int WTN, bool WKMAJ, int EPI>
__global__ __launch_bounds__(WM * WN * 32)
void gemm_wmma(const bf16_t* __restrict__ A, long long az1, long long az2, long long lda,
               const bf16_t* __restrict__ W, long long wz1, long long wz2, long long ldw,
               void* __restrict__ Cv, long long cz1, long long cz2, long long ldc,
               int M, int N, int K, int NZ2,
               const float* __restrict__ bias, const float* __restrict__ scale,
               const float* __restrict__ res, long long resLd,
               const float* __restrict__ aux) {
  constexpr int NT  = WM * WN * 32;
  constexpr int KCH = BK / 8;
  __shared__ alignas(64) bf16_t As[2][BM][BK];
  __shared__ alignas(64) bf16_t Bs[2][BN][BK];
  const int tid = threadIdx.x;
  const int z = blockIdx.z, z1 = z / NZ2, z2 = z % NZ2;
  const long long m0 = (long long)blockIdx.y * BM;
  const long long n0 = (long long)blockIdx.x * BN;
  const bf16_t* Ab = A + z1 * az1 + z2 * az2;
  const bf16_t* Wb = W + z1 * wz1 + z2 * wz2;

  const int wave = tid >> 5, lane = tid & 31;
  const int wm = wave / WN, wn = wave % WN;
  const int lm  = lane & 15;
  const int lkA = (lane >> 4) << 4;   // 0 | 16
  const int lkB = (lane >> 4) << 4;   // 0 | 16

  v8f acc[WTM][WTN];
#pragma unroll
  for (int i = 0; i < WTM; ++i)
#pragma unroll
    for (int j = 0; j < WTN; ++j)
#pragma unroll
      for (int v = 0; v < 8; ++v) acc[i][j][v] = 0.f;

  const int nk = K / BK;

  if constexpr (WKMAJ) {
    // ------------- async double-buffered pipeline (ASYNCcnt) ---------------
    constexpr int APT = (BM * KCH) / NT;
    constexpr int BPT = (BN * KCH) / NT;
    auto stage = [&](int kt, int p) {
      const long long k0 = (long long)kt * BK;
#pragma unroll
      for (int u = 0; u < APT; ++u) {
        int c = tid + u * NT;
        int r = c / KCH, kc = (c % KCH) * 8;
        async_copy_b128(&Ab[(m0 + r) * lda + k0 + kc], &As[p][r][kc]);
      }
#pragma unroll
      for (int u = 0; u < BPT; ++u) {
        int c = tid + u * NT;
        int n = c / KCH, kc = (c % KCH) * 8;
        async_copy_b128(&Wb[(n0 + n) * ldw + k0 + kc], &Bs[p][n][kc]);
      }
    };
    stage(0, 0);
    async_wait0();
    __syncthreads();
    int p = 0;
    for (int kt = 0; kt < nk; ++kt) {
      if (kt + 1 < nk) stage(kt + 1, p ^ 1);
      if (kt + 2 < nk) {
        __builtin_prefetch(&Ab[(m0 + (tid & (BM - 1))) * lda + (long long)(kt + 2) * BK], 0, 1);
        __builtin_prefetch(&Wb[(n0 + (tid & (BN - 1))) * ldw + (long long)(kt + 2) * BK], 0, 1);
      }
      v16bf afr[WTM], bfr[WTN];
#pragma unroll
      for (int i = 0; i < WTM; ++i)
        afr[i] = load_a_frag(&As[p][(wm * WTM + i) * 16 + lm][lkA]);
#pragma unroll
      for (int j = 0; j < WTN; ++j)
        bfr[j] = *(const v16bf*)&Bs[p][(wn * WTN + j) * 16 + lm][lkB];
#pragma unroll
      for (int i = 0; i < WTM; ++i)
#pragma unroll
        for (int j = 0; j < WTN; ++j)
          acc[i][j] = __builtin_amdgcn_wmma_f32_16x16x32_bf16(
              false, afr[i], false, bfr[j], (short)0, acc[i][j], false, false);
      if (kt + 1 < nk) async_wait0();
      __syncthreads();
      p ^= 1;
    }
  } else {
    // ------------- synchronous path (N-major W gather) ---------------------
    for (int kt = 0; kt < nk; ++kt) {
      const long long k0 = (long long)kt * BK;
      for (int c = tid; c < BM * KCH; c += NT) {
        int r = c / KCH, kc = (c % KCH) * 8;
        *(v8bf*)&As[0][r][kc] = *(const v8bf*)&Ab[(m0 + r) * lda + k0 + kc];
      }
      for (int c = tid; c < BN * BK; c += NT) {
        int n = c / BK, kk = c % BK;
        Bs[0][n][kk] = Wb[(long long)(k0 + kk) * ldw + n0 + n];
      }
      __syncthreads();
      v16bf afr[WTM], bfr[WTN];
#pragma unroll
      for (int i = 0; i < WTM; ++i)
        afr[i] = load_a_frag(&As[0][(wm * WTM + i) * 16 + lm][lkA]);
#pragma unroll
      for (int j = 0; j < WTN; ++j)
        bfr[j] = *(const v16bf*)&Bs[0][(wn * WTN + j) * 16 + lm][lkB];
#pragma unroll
      for (int i = 0; i < WTM; ++i)
#pragma unroll
        for (int j = 0; j < WTN; ++j)
          acc[i][j] = __builtin_amdgcn_wmma_f32_16x16x32_bf16(
              false, afr[i], false, bfr[j], (short)0, acc[i][j], false, false);
      __syncthreads();
    }
  }

  // ------------------------------ epilogue ---------------------------------
  float*  Cf = (float*)Cv;
  bf16_t* Chh = (bf16_t*)Cv;
  const long long cb = z1 * cz1 + z2 * cz2;
  const int rofs = (lane >> 4) << 3;  // 0 | 8
#pragma unroll
  for (int i = 0; i < WTM; ++i) {
#pragma unroll
    for (int j = 0; j < WTN; ++j) {
#pragma unroll
      for (int v = 0; v < 8; ++v) {
        long long mrow = m0 + (wm * WTM + i) * 16 + v + rofs;
        long long ncol = n0 + (wn * WTN + j) * 16 + lm;
        if (mrow >= M || ncol >= N) continue;
        float val = acc[i][j][v];
        long long co = cb + mrow * ldc + ncol;
        if constexpr (EPI == EPI_BF16) {
          Chh[co] = f2bf(val);
        } else if constexpr (EPI == EPI_BIAS_BF16) {
          Chh[co] = f2bf(val + bias[ncol]);
        } else if constexpr (EPI == EPI_ACC_F32) {
          Cf[co] += val;
        } else if constexpr (EPI == EPI_BIAS_LS_F32) {
          Cf[co] = (val + bias[ncol]) * scale[ncol];
        } else if constexpr (EPI == EPI_GATE) {
          float gt = sigmoidf_(val + bias[ncol]);
          long long bb = mrow >> 13;          // / KT
          long long t  = mrow & (KT - 1);
          float oup = aux[(bb * KD + (t >> 3)) * KC + ncol];
          Cf[co] = res[mrow * resLd + ncol] + gt * oup;
        } else if constexpr (EPI == EPI_RES_BIAS_LS_F32) {
          Cf[co] = res[mrow * resLd + ncol] + (val + bias[ncol]) * scale[ncol];
        }
      }
    }
  }
}

// ---------------------------------------------------------------------------
// Relative-position scores: for each s, R[bh, t] = Q[bh, s, :] . pos_k[s, t, :]
// grid (KD/128, KD);  M = 32 (bh), N-tile = 128, K = 64.
// ---------------------------------------------------------------------------
__global__ __launch_bounds__(256)
void score_rel_kernel(const bf16_t* __restrict__ q, const bf16_t* __restrict__ posk,
                      float* __restrict__ scores) {
  const int s  = blockIdx.y;
  const int t0 = blockIdx.x * 128;
  __shared__ alignas(64) bf16_t Qs[32][64];
  __shared__ alignas(64) bf16_t Ps[128][64];
  const int tid = threadIdx.x;
  {
    int m = tid >> 3, kc = (tid & 7) * 8;   // 32 rows x 8 chunks
    long long off = (long long)(m >> 3) * (KD * KC) + (long long)s * KC + (m & 7) * KDK + kc;
    async_copy_b128(&q[off], &Qs[m][kc]);
  }
#pragma unroll
  for (int u = 0; u < 4; ++u) {
    int c = threadIdx.x + u * 256;
    int n = c >> 3, kc = (c & 7) * 8;
    async_copy_b128(&posk[((long long)s * KD + t0 + n) * KDK + kc], &Ps[n][kc]);
  }
  async_wait0();
  __syncthreads();
  const int wave = tid >> 5, lane = tid & 31;
  const int wm = wave >> 2, wn = wave & 3;   // 2 x 4 waves
  const int lm = lane & 15, lkA = (lane >> 4) << 4, lkB = (lane >> 4) << 4;
  v8f acc[2];
#pragma unroll
  for (int j = 0; j < 2; ++j)
#pragma unroll
    for (int v = 0; v < 8; ++v) acc[j][v] = 0.f;
#pragma unroll
  for (int ks = 0; ks < 64; ks += 32) {
    v16bf a = load_a_frag(&Qs[wm * 16 + lm][ks + lkA]);
#pragma unroll
    for (int j = 0; j < 2; ++j) {
      v16bf b = *(const v16bf*)&Ps[(wn * 2 + j) * 16 + lm][ks + lkB];
      acc[j] = __builtin_amdgcn_wmma_f32_16x16x32_bf16(
          false, a, false, b, (short)0, acc[j], false, false);
    }
  }
  const int rofs = (lane >> 4) << 3;
#pragma unroll
  for (int j = 0; j < 2; ++j)
#pragma unroll
    for (int v = 0; v < 8; ++v) {
      int m = wm * 16 + v + rofs;                 // bh
      int t = t0 + (wn * 2 + j) * 16 + lm;
      scores[(long long)m * (KD * KD) + (long long)s * KD + t] = acc[j][v];
    }
}

// ---------------------------------------------------------------------------
extern "C" void kernel_launch(void* const* d_in, const int* in_sizes, int n_in,
                              void* d_out, int out_size, void* d_ws, size_t ws_size,
                              hipStream_t stream) {
  const float* x         = (const float*)d_in[0];
  const float* pos_k     = (const float*)d_in[1];
  const float* ln_attn_g = (const float*)d_in[2];
  const float* ln_attn_b = (const float*)d_in[3];
  const float* wq        = (const float*)d_in[4];
  const float* bqv       = (const float*)d_in[5];
  const float* wk        = (const float*)d_in[6];
  const float* bkv       = (const float*)d_in[7];
  const float* wv        = (const float*)d_in[8];
  const float* bvv       = (const float*)d_in[9];
  const float* wo        = (const float*)d_in[10];
  const float* bov       = (const float*)d_in[11];
  const float* ls_attn   = (const float*)d_in[12];
  const float* ln_gate_g = (const float*)d_in[13];
  const float* ln_gate_b = (const float*)d_in[14];
  const float* w_gate    = (const float*)d_in[15];
  const float* b_gate    = (const float*)d_in[16];
  const float* ln_ffn_g  = (const float*)d_in[17];
  const float* ln_ffn_b  = (const float*)d_in[18];
  const float* w1        = (const float*)d_in[19];
  const float* b1        = (const float*)d_in[20];
  const float* dw_w      = (const float*)d_in[21];
  const float* dw_b      = (const float*)d_in[22];
  const float* w2        = (const float*)d_in[23];
  const float* b2        = (const float*)d_in[24];
  const float* ls_ffn    = (const float*)d_in[25];
  float* out = (float*)d_out;

  // ----------------------------- workspace ---------------------------------
  size_t off = 0;
  char* wsb = (char*)d_ws;
  auto alloc = [&](size_t bytes) -> void* {
    void* p = wsb + off;
    off += (bytes + 255) & ~(size_t)255;
    return p;
  };
  bf16_t* wq_b  = (bf16_t*)alloc((size_t)KC * KC * 2);
  bf16_t* wk_b  = (bf16_t*)alloc((size_t)KC * KC * 2);
  bf16_t* wv_b  = (bf16_t*)alloc((size_t)KC * KC * 2);
  bf16_t* wo_b  = (bf16_t*)alloc((size_t)KC * KC * 2);
  bf16_t* wg_b  = (bf16_t*)alloc((size_t)KC * KC * 2);
  bf16_t* w1_b  = (bf16_t*)alloc((size_t)K6C * KC * 2);
  bf16_t* w2_b  = (bf16_t*)alloc((size_t)KC * K3C * 2);
  bf16_t* pos_b = (bf16_t*)alloc((size_t)KD * KD * KDK * 2);   // 128 MB -> lives in L2
  float*  xt    = (float*) alloc((size_t)KB * KT * KC * 4);    // (B,T,C)
  bf16_t* hbuf  = (bf16_t*)alloc((size_t)KB * KD * KC * 2);    // LN(pooled)
  bf16_t* qbuf  = (bf16_t*)alloc((size_t)KB * KD * KC * 2);
  bf16_t* kbuf  = (bf16_t*)alloc((size_t)KB * KD * KC * 2);
  bf16_t* vbuf  = (bf16_t*)alloc((size_t)KB * KD * KC * 2);
  float*  scores= (float*) alloc((size_t)KB * KH * KD * KD * 4); // attn aliased in-place
  bf16_t* ohead = (bf16_t*)alloc((size_t)KB * KD * KC * 2);
  float*  oproj = (float*) alloc((size_t)KB * KD * KC * 4);
  bf16_t* xtln  = (bf16_t*)alloc((size_t)KB * KT * KC * 2);    // reused for LN(y)
  float*  ybuf  = (float*) alloc((size_t)KB * KT * KC * 4);
  bf16_t* zbuf  = (bf16_t*)alloc((size_t)KB * KT * K6C * 2);
  bf16_t* gbuf  = (bf16_t*)alloc((size_t)KB * KT * K3C * 2);
  (void)ws_size; (void)n_in; (void)in_sizes; (void)out_size;

  // -------------------- 1) fp32 -> bf16 weight/pos_k conversion ------------
  auto cvt = [&](const float* src, bf16_t* dst, long long n) {
    int blocks = (int)((n + 256LL * 8 - 1) / (256LL * 8));
    if (blocks > 65535) blocks = 65535;
    cvt_bf16_kernel<<<blocks, 256, 0, stream>>>(src, dst, n);
  };
  cvt(wq, wq_b, (long long)KC * KC);
  cvt(wk, wk_b, (long long)KC * KC);
  cvt(wv, wv_b, (long long)KC * KC);
  cvt(wo, wo_b, (long long)KC * KC);
  cvt(w_gate, wg_b, (long long)KC * KC);
  cvt(w1, w1_b, (long long)K6C * KC);
  cvt(w2, w2_b, (long long)KC * K3C);
  cvt(pos_k, pos_b, (long long)KD * KD * KDK);

  // -------------------- 2) x (B,C,T) -> xt (B,T,C) -------------------------
  {
    dim3 g(KT / 32, KC / 32, KB), b(32, 8);
    transpose_kernel<<<g, b, 0, stream>>>(x, xt, KC, KT);
  }

  // -------------------- 3) pool(8) + LN_attn -> h bf16 ---------------------
  pool_ln_kernel<<<KB * KD, 256, 0, stream>>>(xt, ln_attn_g, ln_attn_b, hbuf);

  // -------------------- 4) Q/K/V projections (WMMA, async pipeline) --------
  {
    dim3 g(KC / 128, (KB * KD) / 128, 1);
    gemm_wmma<128,128,32, 2,4, 4,2, true, EPI_BIAS_BF16><<<g, 256, 0, stream>>>(
        hbuf, 0, 0, KC, wq_b, 0, 0, KC, qbuf, 0, 0, KC,
        KB * KD, KC, KC, 1, bqv, nullptr, nullptr, 0, nullptr);
    gemm_wmma<128,128,32, 2,4, 4,2, true, EPI_BIAS_BF16><<<g, 256, 0, stream>>>(
        hbuf, 0, 0, KC, wk_b, 0, 0, KC, kbuf, 0, 0, KC,
        KB * KD, KC, KC, 1, bkv, nullptr, nullptr, 0, nullptr);
    gemm_wmma<128,128,32, 2,4, 4,2, true, EPI_BIAS_BF16><<<g, 256, 0, stream>>>(
        hbuf, 0, 0, KC, wv_b, 0, 0, KC, vbuf, 0, 0, KC,
        KB * KD, KC, KC, 1, bvv, nullptr, nullptr, 0, nullptr);
  }

  // -------------------- 5) rel-pos scores R (WMMA, per-s GEMM) -------------
  {
    dim3 g(KD / 128, KD, 1);
    score_rel_kernel<<<g, 256, 0, stream>>>(qbuf, pos_b, scores);
  }

  // -------------------- 6) content scores  S += Q.K^T ----------------------
  {
    dim3 g(KD / 128, KD / 128, KB * KH);
    gemm_wmma<128,128,32, 2,4, 4,2, true, EPI_ACC_F32><<<g, 256, 0, stream>>>(
        qbuf, (long long)KD * KC, KDK, KC,
        kbuf, (long long)KD * KC, KDK, KC,
        scores, (long long)KH * KD * KD, (long long)KD * KD, KD,
        KD, KD, KDK, KH, nullptr, nullptr, nullptr, 0, nullptr);
  }

  // -------------------- 7) softmax((A+R)/8) -> attn bf16 (in-place) --------
  softmax_kernel<<<KB * KH * KD, 256, 0, stream>>>(scores);

  // -------------------- 8) o = attn . V (WMMA, N-major W) ------------------
  {
    const bf16_t* attn = (const bf16_t*)scores;  // row stride 2048 bf16
    dim3 g(1, KD / 256, KB * KH);
    gemm_wmma<256,64,32, 4,2, 4,2, false, EPI_BF16><<<g, 256, 0, stream>>>(
        attn, (long long)KH * 2048 * KD, (long long)2048 * KD, 2048,
        vbuf, (long long)KD * KC, KDK, KC,
        ohead, (long long)KD * KC, KDK, KC,
        KD, KDK, KD, KH, nullptr, nullptr, nullptr, 0, nullptr);
  }

  // -------------------- 9) wo projection + bias + LayerScale ---------------
  {
    dim3 g(KC / 128, (KB * KD) / 128, 1);
    gemm_wmma<128,128,32, 2,4, 4,2, true, EPI_BIAS_LS_F32><<<g, 256, 0, stream>>>(
        ohead, 0, 0, KC, wo_b, 0, 0, KC, oproj, 0, 0, KC,
        KB * KD, KC, KC, 1, bov, ls_attn, nullptr, 0, nullptr);
  }

  // -------------------- 10) LN_gate(xt) -> bf16 ----------------------------
  rowln_kernel<<<KB * KT, 256, 0, stream>>>(xt, ln_gate_g, ln_gate_b, xtln);

  // -------------------- 11) gate GEMM + sigmoid + residual + upsample ------
  {
    dim3 g(KC / 128, (KB * KT) / 128, 1);
    gemm_wmma<128,128,32, 2,4, 4,2, true, EPI_GATE><<<g, 256, 0, stream>>>(
        xtln, 0, 0, KC, wg_b, 0, 0, KC, ybuf, 0, 0, KC,
        KB * KT, KC, KC, 1, b_gate, nullptr, xt, KC, oproj);
  }

  // -------------------- 12) LN_ffn(y) -> bf16 (reuse xtln) -----------------
  rowln_kernel<<<KB * KT, 256, 0, stream>>>(ybuf, ln_ffn_g, ln_ffn_b, xtln);

  // -------------------- 13) ffn1: z = LN(y).w1^T + b1 ----------------------
  {
    dim3 g(K6C / 128, (KB * KT) / 128, 1);
    gemm_wmma<128,128,32, 2,4, 4,2, true, EPI_BIAS_BF16><<<g, 256, 0, stream>>>(
        xtln, 0, 0, KC, w1_b, 0, 0, KC, zbuf, 0, 0, K6C,
        KB * KT, K6C, KC, 1, b1, nullptr, nullptr, 0, nullptr);
  }

  // -------------------- 14) depthwise conv(3) + GLU -> g -------------------
  {
    long long total = (long long)KB * KT * (K3C / 8);
    int blocks = (int)((total + 255) / 256);
    conv_glu_kernel<<<blocks, 256, 0, stream>>>(zbuf, dw_w, dw_b, gbuf);
  }

  // -------------------- 15) ffn2 + bias*ls + residual (in-place y) ---------
  {
    dim3 g(KC / 128, (KB * KT) / 128, 1);
    gemm_wmma<128,128,32, 2,4, 4,2, true, EPI_RES_BIAS_LS_F32><<<g, 256, 0, stream>>>(
        gbuf, 0, 0, K3C, w2_b, 0, 0, K3C, ybuf, 0, 0, KC,
        KB * KT, KC, K3C, 1, b2, ls_ffn, ybuf, KC, nullptr);
  }

  // -------------------- 16) y (B,T,C) -> out (B,C,T) -----------------------
  {
    dim3 g(KC / 32, KT / 32, KB), b(32, 8);
    transpose_kernel<<<g, b, 0, stream>>>(ybuf, out, KT, KC);
  }
}
